// GatedAttention_11922829214183
// MI455X (gfx1250) — compile-verified, matrix operand order checked by reading, not run on hardware
//
#include <hip/hip_runtime.h>
#include <hip/hip_bf16.h>

typedef __bf16 bf16_t;
typedef __attribute__((ext_vector_type(2)))  __bf16 v2bf;
typedef __attribute__((ext_vector_type(4)))  __bf16 v4bf;
typedef __attribute__((ext_vector_type(8)))  __bf16 v8bf;
typedef __attribute__((ext_vector_type(16))) __bf16 v16bf;
typedef __attribute__((ext_vector_type(8)))  float  v8f;

#define HQ_  16
#define HKV_ 2
#define D_   128
#define G_   8

// Swizzled B-tile layout (matches ISA 16-bit B fragment layout):
// per 32(K)x16(N) subtile of 512 bf16: lane = ((k>>3)&1)*16 + (n&15),
// elem = ((k>>4)&1)*8 + (k&7).  Fragment read = 32 contiguous bytes / lane.
__device__ __forceinline__ int bswz(int k, int n, int ntiles) {
  int kc = k >> 5, nt = n >> 4;
  int lane = (((k >> 3) & 1) << 4) + (n & 15);
  int elem = (((k >> 4) & 1) << 3) + (k & 7);
  return ((kc * ntiles + nt) << 9) + (lane << 4) + elem;
}

__device__ __forceinline__ v16bf cat8(v8bf a0, v8bf a1) {
  return __builtin_shufflevector(a0, a1, 0, 1, 2, 3, 4, 5, 6, 7,
                                 8, 9, 10, 11, 12, 13, 14, 15);
}

// ---------------------------------------------------------------------------
// Tiled GEMM: C(MxN) = A(MxK) * B(KxN). Block tile 128x64, BK=64, 8 waves,
// each wave 32x32 via 4 WMMA tiles per 32-K chunk.
// ---------------------------------------------------------------------------
template <typename AT, bool OUT_BF16>
__global__ __launch_bounds__(256) void gemm_wmma(
    const AT* __restrict__ A, const float* __restrict__ Bw,
    void* __restrict__ Cout, int M, int N, int K) {
  constexpr int BM = 128, BN = 64, BK = 64;
  __shared__ alignas(32) bf16_t As[BM][BK + 8];                 // pitch 72 (144B)
  __shared__ alignas(32) bf16_t Bsw[(BK / 32) * (BN / 16) * 512];

  const int tid  = threadIdx.x;
  const int lane = tid & 31;
  const int wave = tid >> 5;
  const int wm   = (wave & 3) * 32;
  const int wn   = (wave >> 2) * 32;
  const int bm   = blockIdx.x * BM;
  const int bn   = blockIdx.y * BN;
  const int kh   = (lane >> 4) * 8;
  const int lr   = lane & 15;

  v8f acc[2][2] = {};

  for (int k0 = 0; k0 < K; k0 += BK) {
    // A fill: pairs along K (contiguous global + v2bf LDS store).
    for (int i = tid; i < BM * (BK / 2); i += 256) {
      int r = i >> 5, c = (i & 31) * 2;
      int gr = bm + r;
      v2bf pk;
      if (gr < M) {
        const AT* ap = A + (size_t)gr * K + (k0 + c);
        pk[0] = (bf16_t)(float)ap[0];
        pk[1] = (bf16_t)(float)ap[1];
      } else { pk[0] = (bf16_t)0.f; pk[1] = (bf16_t)0.f; }
      *(v2bf*)&As[r][c] = pk;
    }
    // B fill: pairs along K into swizzled layout (coalesced rows of Bw).
    for (int i = tid; i < (BK / 2) * BN; i += 256) {
      int n = i & 63, k = (i >> 6) * 2;
      int gc = bn + n;
      float v0 = 0.f, v1 = 0.f;
      if (gc < N) {
        v0 = Bw[(size_t)(k0 + k) * N + gc];
        v1 = Bw[(size_t)(k0 + k + 1) * N + gc];
      }
      v2bf pk; pk[0] = (bf16_t)v0; pk[1] = (bf16_t)v1;
      *(v2bf*)&Bsw[bswz(k, n, BN / 16)] = pk;
    }
    __syncthreads();

#pragma unroll
    for (int kc = 0; kc < BK / 32; ++kc) {
#pragma unroll
      for (int mt = 0; mt < 2; ++mt) {
        const bf16_t* ap = &As[wm + mt * 16 + lr][kc * 32];
        v16bf a = cat8(*(const v8bf*)(ap + kh), *(const v8bf*)(ap + 16 + kh));
#pragma unroll
        for (int nt = 0; nt < 2; ++nt) {
          int nti = (wn >> 4) + nt;
          v16bf b = *(const v16bf*)&Bsw[((kc * (BN / 16) + nti) << 9) + (lane << 4)];
          acc[mt][nt] = __builtin_amdgcn_wmma_f32_16x16x32_bf16(
              false, a, false, b, (short)0, acc[mt][nt], false, false);
        }
      }
    }
    __syncthreads();
  }

  const int crow0 = (lane >> 4) * 8;
  const int ccol  = lane & 15;
#pragma unroll
  for (int mt = 0; mt < 2; ++mt)
#pragma unroll
    for (int nt = 0; nt < 2; ++nt) {
      int gn = bn + wn + nt * 16 + ccol;
      if (gn >= N) continue;
      int gm0 = bm + wm + mt * 16 + crow0;
#pragma unroll
      for (int r = 0; r < 8; ++r) {
        int gm = gm0 + r;
        if (gm < M) {
          if (OUT_BF16) ((bf16_t*)Cout)[(size_t)gm * N + gn] = (bf16_t)acc[mt][nt][r];
          else          ((float*)Cout)[(size_t)gm * N + gn]  = acc[mt][nt][r];
        }
      }
    }
}

// ---------------------------------------------------------------------------
// Flash attention, causal, GQA, sigmoid gate in epilogue.
// 4 waves / block; 64 queries x D=128 per block.
// ---------------------------------------------------------------------------
__global__ __launch_bounds__(128) void attn_kernel(
    const bf16_t* __restrict__ Q, const bf16_t* __restrict__ Km,
    const bf16_t* __restrict__ Vm, const float* __restrict__ Glogit,
    bf16_t* __restrict__ Oo, int T) {
  __shared__ alignas(32) bf16_t Qs[64][D_ + 8];          // pitch 136 (272B)
  __shared__ alignas(32) bf16_t Ktsw[4 * 4 * 512];       // K^T: k=d(128), n=key(64)
  __shared__ alignas(32) bf16_t Vsw[2 * 8 * 512];        // V:   k=key(64), n=d(128)
  __shared__ alignas(16) float  Ss[64][64 + 4];          // pitch 68 (272B)
  __shared__ alignas(32) bf16_t Ps[64][64 + 8];          // pitch 72 (144B)
  __shared__ float rowM[64], rowL[64], rowA[64];

  const int tid   = threadIdx.x;
  const int lane  = tid & 31;
  const int w     = tid >> 5;
  const int lr    = lane & 15;
  const int kh    = (lane >> 4) * 8;
  const int crow0 = (lane >> 4) * 8;
  const int ccol  = lane & 15;

  const int qtile = blockIdx.x;
  const int hq    = blockIdx.y;
  const int bz    = blockIdx.z;
  const int kv    = hq / G_;
  const int qbase = qtile * 64;
  const float scale = 0.08838834764831845f;  // 1/sqrt(128)

  // Q tile: v2bf copies (coalesced).
  for (int i = tid; i < 64 * (D_ / 2); i += 128) {
    int r = i >> 6, dp = (i & 63) * 2;
    *(v2bf*)&Qs[r][dp] =
        *(const v2bf*)&Q[((size_t)(bz * T + qbase + r)) * (HQ_ * D_) + hq * D_ + dp];
  }
  { int rr = w * 16 + lr; rowM[rr] = -3.0e38f; rowL[rr] = 0.0f; }
  __syncthreads();

  v8f o[8] = {};

  for (int kt = 0; kt <= qtile; ++kt) {
    const int kb = kt * 64;
    const size_t krow0 = ((size_t)(bz * T + kb)) * (HKV_ * D_) + kv * D_;
    // K^T fill: pairs along d (== K dim of S-matmul), coalesced v2bf loads.
    for (int i = tid; i < 64 * (D_ / 2); i += 128) {
      int key = i >> 6, dp = (i & 63) * 2;
      v2bf kvp = *(const v2bf*)&Km[krow0 + (size_t)key * (HKV_ * D_) + dp];
      *(v2bf*)&Ktsw[bswz(dp, key, 4)] = kvp;
    }
    // V fill: pairs along key (== K dim of PV-matmul), coalesced u16 loads.
    for (int i = tid; i < 32 * D_; i += 128) {
      int d = i & (D_ - 1), kp = (i >> 7) * 2;
      v2bf pk;
      pk[0] = Vm[krow0 + (size_t)kp * (HKV_ * D_) + d];
      pk[1] = Vm[krow0 + (size_t)(kp + 1) * (HKV_ * D_) + d];
      *(v2bf*)&Vsw[bswz(kp, d, 8)] = pk;
    }
    __syncthreads();

    // ---- S = Q * K^T ----
    v8f s[4] = {};
#pragma unroll
    for (int d0 = 0; d0 < D_; d0 += 32) {
      const bf16_t* ap = &Qs[w * 16 + lr][d0];
      v16bf a = cat8(*(const v8bf*)(ap + kh), *(const v8bf*)(ap + 16 + kh));
      int kc = d0 >> 5;
#pragma unroll
      for (int nt = 0; nt < 4; ++nt) {
        v16bf b = *(const v16bf*)&Ktsw[((kc * 4 + nt) << 9) + (lane << 4)];
        s[nt] = __builtin_amdgcn_wmma_f32_16x16x32_bf16(
            false, a, false, b, (short)0, s[nt], false, false);
      }
    }

    // Scale + causal mask, spill to LDS.
#pragma unroll
    for (int nt = 0; nt < 4; ++nt) {
      int key = kb + nt * 16 + ccol;
#pragma unroll
      for (int r = 0; r < 8; ++r) {
        int mrow = crow0 + r;
        float v = s[nt][r] * scale;
        if (key > qbase + w * 16 + mrow) v = -1.0e30f;
        Ss[w * 16 + mrow][nt * 16 + ccol] = v;
      }
    }

    // ---- online softmax (one row per lane-pair; vectorized LDS) ----
    {
      int rr = w * 16 + lr;
      const float4* srow = (const float4*)&Ss[rr][0];
      float mold = rowM[rr];
      float mnew = mold;
#pragma unroll
      for (int jj = 0; jj < 16; ++jj) {
        float4 s4 = srow[jj];
        mnew = fmaxf(mnew, fmaxf(fmaxf(s4.x, s4.y), fmaxf(s4.z, s4.w)));
      }
      float alpha = __expf(mold - mnew);
      float l = rowL[rr] * alpha;
#pragma unroll
      for (int jj = 0; jj < 16; ++jj) {
        float4 s4 = srow[jj];
        float p0 = __expf(s4.x - mnew), p1 = __expf(s4.y - mnew);
        float p2 = __expf(s4.z - mnew), p3 = __expf(s4.w - mnew);
        l += (p0 + p1) + (p2 + p3);
        v4bf pv; pv[0] = (bf16_t)p0; pv[1] = (bf16_t)p1;
        pv[2] = (bf16_t)p2; pv[3] = (bf16_t)p3;
        *(v4bf*)&Ps[rr][jj * 4] = pv;
      }
      rowM[rr] = mnew; rowL[rr] = l; rowA[rr] = alpha;
    }

    // Rescale accumulator by alpha(row).
    float al[8];
#pragma unroll
    for (int r = 0; r < 8; ++r) al[r] = rowA[w * 16 + crow0 + r];
#pragma unroll
    for (int dt = 0; dt < 8; ++dt)
#pragma unroll
      for (int r = 0; r < 8; ++r) o[dt][r] *= al[r];

    // ---- O += P * V ----
#pragma unroll
    for (int kk = 0; kk < 64; kk += 32) {
      const bf16_t* ap = &Ps[w * 16 + lr][kk];
      v16bf a = cat8(*(const v8bf*)(ap + kh), *(const v8bf*)(ap + 16 + kh));
      int kc = kk >> 5;
#pragma unroll
      for (int dt = 0; dt < 8; ++dt) {
        v16bf b = *(const v16bf*)&Vsw[((kc * 8 + dt) << 9) + (lane << 4)];
        o[dt] = __builtin_amdgcn_wmma_f32_16x16x32_bf16(
            false, a, false, b, (short)0, o[dt], false, false);
      }
    }
    __syncthreads();
  }

  // Epilogue: 1/l, sigmoid gate, store bf16.
  float g[8], il[8];
#pragma unroll
  for (int r = 0; r < 8; ++r) {
    int rr = w * 16 + crow0 + r;
    float lg = Glogit[((size_t)(bz * T + qbase + rr)) * HQ_ + hq];
    g[r]  = 1.0f / (1.0f + __expf(-lg));
    il[r] = 1.0f / rowL[rr];
  }
#pragma unroll
  for (int dt = 0; dt < 8; ++dt) {
    int d = dt * 16 + ccol;
#pragma unroll
    for (int r = 0; r < 8; ++r) {
      int q = qbase + w * 16 + crow0 + r;
      Oo[((size_t)(bz * T + q)) * (HQ_ * D_) + hq * D_ + d] =
          (bf16_t)(o[dt][r] * g[r] * il[r]);
    }
  }
}

// ---------------------------------------------------------------------------
extern "C" void kernel_launch(void* const* d_in, const int* in_sizes, int n_in,
                              void* d_out, int out_size, void* d_ws, size_t ws_size,
                              hipStream_t stream) {
  const int B = 2, T = 2048, C = 2048;
  const int BT = B * T;

  const float* x  = (const float*)d_in[0];
  const float* Wq = (const float*)d_in[2];
  const float* Wk = (const float*)d_in[3];
  const float* Wv = (const float*)d_in[4];
  const float* Wo = (const float*)d_in[5];
  const float* Wg = (const float*)d_in[6];
  float* out = (float*)d_out;

  bf16_t* Qb  = (bf16_t*)d_ws;
  bf16_t* Kb  = Qb + (size_t)BT * (HQ_ * D_);
  bf16_t* Vb  = Kb + (size_t)BT * (HKV_ * D_);
  float*  Gl  = (float*)(Vb + (size_t)BT * (HKV_ * D_));
  bf16_t* AOb = (bf16_t*)(Gl + (size_t)BT * HQ_);

  dim3 blk(256);
  gemm_wmma<float, true><<<dim3(BT / 128, (HQ_ * D_) / 64), blk, 0, stream>>>(
      x, Wq, Qb, BT, HQ_ * D_, C);
  gemm_wmma<float, true><<<dim3(BT / 128, (HKV_ * D_) / 64), blk, 0, stream>>>(
      x, Wk, Kb, BT, HKV_ * D_, C);
  gemm_wmma<float, true><<<dim3(BT / 128, (HKV_ * D_) / 64), blk, 0, stream>>>(
      x, Wv, Vb, BT, HKV_ * D_, C);
  gemm_wmma<float, false><<<dim3(BT / 128, 1), blk, 0, stream>>>(
      x, Wg, Gl, BT, HQ_, C);

  attn_kernel<<<dim3(T / 64, HQ_, B), dim3(128), 0, stream>>>(
      Qb, Kb, Vb, Gl, AOb, T);

  gemm_wmma<bf16_t, false><<<dim3(BT / 128, C / 64), blk, 0, stream>>>(
      AOb, Wo, out, BT, C, HQ_ * D_);
}
